// Head_12017318494627
// MI455X (gfx1250) — matrix operand order repaired and checked by reading, not compiled, not run on hardware
//
#include <hip/hip_runtime.h>
#include <hip/hip_bf16.h>

// ME-GraphAU forward for MI455X (gfx1250, wave32).
// All GEMMs run on v_wmma_f32_16x16x32_bf16 (bf16 inputs, fp32 accum), register-
// blocked 32x64 per wave (2x4 subtiles, 8 WMMAs per K-step) for ~21 FLOP/B from L2.
// Softmax / BatchNorm / sigmoid stay in fp32 scalar VALU (tiny FLOPs, numerically
// sensitive). Full `edge` tensor (231MB fp32) is never materialized -- BN+mean
// collapse analytically. Workspace use: ~370 MB.

#define B_  16
#define N_  49
#define C_  512
#define NC_ 12
#define EPS_ 1e-5f

typedef __bf16 bf16;
typedef __bf16 v16bf __attribute__((ext_vector_type(16)));
typedef float  v8f   __attribute__((ext_vector_type(8)));

// ---------------------------------------------------------------------------
// WMMA GEMM:  C(MxN) = A(MxK) @ W(NxK)^T  (+bias per col) (+relu)
// A, W row-major bf16.  Requires N%64==0, K%32==0; M arbitrary multiple of 16
// (A-row reads clamped, stores guarded).  One wave per 32x64 macro-tile:
// 2 A-fragments x 4 B-fragments -> 8 v_wmma per 32-wide K-step.
// ---------------------------------------------------------------------------
__global__ void __launch_bounds__(32)
k_gemm_wmma(const bf16* __restrict__ A, const bf16* __restrict__ W,
            const float* __restrict__ bias,
            float* __restrict__ Cf, bf16* __restrict__ Cb,
            int M, int N, int K, int relu)
{
    const int tilesN = N >> 6;              // 64-column macro tiles
    const int tile   = blockIdx.x;
    const int tn     = tile % tilesN;
    const int tm     = tile / tilesN;
    const int lane   = threadIdx.x;
    const int half   = lane >> 4;           // 0: K-low group, 1: K-high group
    const int l16    = lane & 15;

    // A: lane l16 holds row M=l16 (both lane-halves), K split {0..7,16..23}/{8..15,24..31}
    const bf16* arow[2];
#pragma unroll
    for (int mt = 0; mt < 2; ++mt) {
        int r = tm * 32 + mt * 16 + l16;
        if (r > M - 1) r = M - 1;           // clamp: keep reads in-bounds, never stored
        arow[mt] = A + (size_t)r * K;
    }
    // B: lane l16 holds col N=l16, K split {0..7,16..23}/{8..15,24..31} (matches A)
    const bf16* wrow[4];
#pragma unroll
    for (int nt = 0; nt < 4; ++nt)
        wrow[nt] = W + (size_t)(tn * 64 + nt * 16 + l16) * K;

    v8f acc[2][4];
#pragma unroll
    for (int mt = 0; mt < 2; ++mt)
#pragma unroll
        for (int nt = 0; nt < 4; ++nt)
            acc[mt][nt] = (v8f){0.f, 0.f, 0.f, 0.f, 0.f, 0.f, 0.f, 0.f};

    for (int k0 = 0; k0 < K; k0 += 32) {
        v16bf a[2], b[4];
        const int abase = k0 + half * 8;
#pragma unroll
        for (int mt = 0; mt < 2; ++mt)
#pragma unroll
            for (int e = 0; e < 8; ++e) {
                a[mt][e]     = arow[mt][abase + e];
                a[mt][e + 8] = arow[mt][abase + 16 + e];
            }
        const int bbase = k0 + half * 8;
#pragma unroll
        for (int nt = 0; nt < 4; ++nt)
#pragma unroll
            for (int e = 0; e < 8; ++e) { b[nt][e] = wrow[nt][bbase + e]; b[nt][e + 8] = wrow[nt][bbase + 16 + e]; }

#pragma unroll
        for (int mt = 0; mt < 2; ++mt)
#pragma unroll
            for (int nt = 0; nt < 4; ++nt)
                acc[mt][nt] = __builtin_amdgcn_wmma_f32_16x16x32_bf16(
                    false, a[mt], false, b[nt], (short)0, acc[mt][nt], false, false);
    }

#pragma unroll
    for (int nt = 0; nt < 4; ++nt) {
        const int col = tn * 64 + nt * 16 + l16;
        const float bv = bias ? bias[col] : 0.0f;
#pragma unroll
        for (int mt = 0; mt < 2; ++mt) {
#pragma unroll
            for (int r = 0; r < 8; ++r) {
                const int row = tm * 32 + mt * 16 + r + half * 8;  // D: VGPR r -> M=r+8*half
                if (row >= M) continue;
                float v = acc[mt][nt][r] + bv;
                if (relu) v = fmaxf(v, 0.0f);
                const size_t idx = (size_t)row * N + col;
                if (Cf) Cf[idx] = v;
                if (Cb) Cb[idx] = (bf16)v;
            }
        }
    }
}

// ---------------------------------------------------------------------------
// Elementwise / reduction helpers
// ---------------------------------------------------------------------------
__global__ void k_cvt(const float* __restrict__ s, bf16* __restrict__ d, size_t n)
{
    size_t i = (size_t)blockIdx.x * blockDim.x + threadIdx.x;
    const size_t stride = (size_t)gridDim.x * blockDim.x;
    for (; i < n; i += stride) d[i] = (bf16)s[i];
}

// per-column batch stats (BN over rows): one thread per column, coalesced
__global__ void k_col_stats(const float* __restrict__ H, float* __restrict__ m,
                            float* __restrict__ rs, int rows, int cols)
{
    const int col = blockIdx.x * blockDim.x + threadIdx.x;
    if (col >= cols) return;
    float s1 = 0.f, s2 = 0.f;
    for (int r = 0; r < rows; ++r) {
        const float v = H[(size_t)r * cols + col];
        s1 += v; s2 += v * v;
    }
    const float mm  = s1 / rows;
    const float var = s2 / rows - mm * mm;
    m[col]  = mm;
    rs[col] = rsqrtf(var + EPS_);
}

__global__ void k_bn_relu_bf16(const float* __restrict__ H, const float* __restrict__ m,
                               const float* __restrict__ rs, bf16* __restrict__ out,
                               size_t n, int cols)
{
    const size_t idx = (size_t)blockIdx.x * blockDim.x + threadIdx.x;
    if (idx >= n) return;
    const int col = (int)(idx % cols);
    const float v = (H[idx] - m[col]) * rs[col];
    out[idx] = (bf16)fmaxf(v, 0.0f);
}

// f_v[b,k,c] = mean_n f_u[(b,n,k),c]   (f_u rows are (b,n,k)-ordered)
__global__ void k_fv(const bf16* __restrict__ fu, float* __restrict__ fv,
                     bf16* __restrict__ fvb)
{
    const int bk = blockIdx.x;            // b*12+k
    const int k  = bk % NC_, b = bk / NC_;
    for (int c = threadIdx.x; c < C_; c += blockDim.x) {
        float s = 0.f;
        for (int n = 0; n < N_; ++n)
            s += (float)fu[(((size_t)(b * N_ + n)) * NC_ + k) * C_ + c];
        const float v = s * (1.0f / N_);
        fv[(size_t)bk * C_ + c]  = v;
        fvb[(size_t)bk * C_ + c] = (bf16)v;
    }
}

// attention stage 1: one block per (b, class k, query n); keys/values from x-projections
__global__ void __launch_bounds__(128)
k_attn1(const float* __restrict__ q1, const float* __restrict__ k1,
        const float* __restrict__ v1, bf16* __restrict__ feat)
{
    __shared__ float qs[256];
    __shared__ float sc[N_];
    __shared__ float invs;
    const int bid = blockIdx.x;
    const int nq = bid % N_;
    const int kc = (bid / N_) % NC_;
    const int b  = bid / (N_ * NC_);

    const float* qrow = q1 + ((size_t)((b * N_ + nq) * NC_ + kc)) * 256;
    for (int c = threadIdx.x; c < 256; c += 128) qs[c] = qrow[c];
    __syncthreads();

    const int j = threadIdx.x;
    if (j < N_) {
        const float* krow = k1 + ((size_t)(b * N_ + j)) * 256;
        float s = 0.f;
        for (int c = 0; c < 256; ++c) s += qs[c] * krow[c];
        sc[j] = s * 0.0625f;                    // (C/2)^-0.5
    }
    __syncthreads();
    if (threadIdx.x == 0) {
        float mx = sc[0];
        for (int t = 1; t < N_; ++t) mx = fmaxf(mx, sc[t]);
        float sum = 0.f;
        for (int t = 0; t < N_; ++t) { const float e = __expf(sc[t] - mx); sc[t] = e; sum += e; }
        invs = 1.0f / sum;
    }
    __syncthreads();
    const float inv = invs;
    bf16* orow = feat + ((size_t)((b * NC_ + kc) * N_ + nq)) * C_;
    for (int c = threadIdx.x; c < C_; c += 128) {
        float acc = 0.f;
        for (int t = 0; t < N_; ++t)
            acc += sc[t] * v1[((size_t)(b * N_ + t)) * C_ + c];
        orow[c] = (bf16)(acc * inv);
    }
}

// attention stage 2: one block per (b, i-class, j-class, query n).
// q from feat[j] projections, k/v from feat[i] projections; writes pair slot i*12+j.
__global__ void __launch_bounds__(128)
k_attn2(const float* __restrict__ q2, const float* __restrict__ k2,
        const float* __restrict__ v2, bf16* __restrict__ feat2)
{
    __shared__ float qs[256];
    __shared__ float sc[N_];
    __shared__ float invs;
    const int bid = blockIdx.x;
    const int nq = bid % N_;
    const int jc = (bid / N_) % NC_;
    const int ic = (bid / (N_ * NC_)) % NC_;
    const int b  = bid / (N_ * NC_ * NC_);

    const float* qrow = q2 + ((size_t)((b * NC_ + jc) * N_ + nq)) * 256;
    for (int c = threadIdx.x; c < 256; c += 128) qs[c] = qrow[c];
    __syncthreads();

    const int j = threadIdx.x;
    if (j < N_) {
        const float* krow = k2 + ((size_t)((b * NC_ + ic) * N_ + j)) * 256;
        float s = 0.f;
        for (int c = 0; c < 256; ++c) s += qs[c] * krow[c];
        sc[j] = s * 0.0625f;
    }
    __syncthreads();
    if (threadIdx.x == 0) {
        float mx = sc[0];
        for (int t = 1; t < N_; ++t) mx = fmaxf(mx, sc[t]);
        float sum = 0.f;
        for (int t = 0; t < N_; ++t) { const float e = __expf(sc[t] - mx); sc[t] = e; sum += e; }
        invs = 1.0f / sum;
    }
    __syncthreads();
    const float inv = invs;
    bf16* orow = feat2 + ((size_t)((b * NC_ * NC_ + ic * NC_ + jc) * N_ + nq)) * C_;
    for (int c = threadIdx.x; c < C_; c += 128) {
        float acc = 0.f;
        for (int t = 0; t < N_; ++t)
            acc += sc[t] * v2[((size_t)((b * NC_ + ic) * N_ + t)) * C_ + c];
        orow[c] = (bf16)(acc * inv);
    }
}

// sum over the N axis of the bf16 ep-projection:  sumN[b,s,c] = sum_n h[b,s,n,c]
__global__ void k_sumN(const bf16* __restrict__ hbf, float* __restrict__ sumN)
{
    const int bs = blockIdx.x;            // b*144+s
    for (int c = threadIdx.x; c < C_; c += blockDim.x) {
        float s = 0.f;
        for (int n = 0; n < N_; ++n)
            s += (float)hbf[((size_t)bs * N_ + n) * C_ + c];
        sumN[(size_t)bs * C_ + c] = s;
    }
}

// per-slot mean/rsqrt(var) over {outer x inner} region (block = slot, 256 threads)
template <typename T>
__global__ void k_slot_stats(const T* __restrict__ X, float* __restrict__ m,
                             float* __restrict__ rs, int nouter, size_t ostride,
                             size_t sstride, int inner, float invcnt)
{
    __shared__ float s1s[256], s2s[256];
    const int s = blockIdx.x;
    float s1 = 0.f, s2 = 0.f;
    for (int o = 0; o < nouter; ++o) {
        const T* base = X + (size_t)o * ostride + (size_t)s * sstride;
        for (int i = threadIdx.x; i < inner; i += 256) {
            const float v = (float)base[i];
            s1 += v; s2 += v * v;
        }
    }
    s1s[threadIdx.x] = s1; s2s[threadIdx.x] = s2;
    __syncthreads();
    for (int w = 128; w > 0; w >>= 1) {
        if ((int)threadIdx.x < w) {
            s1s[threadIdx.x] += s1s[threadIdx.x + w];
            s2s[threadIdx.x] += s2s[threadIdx.x + w];
        }
        __syncthreads();
    }
    if (threadIdx.x == 0) {
        const float mm  = s1s[0] * invcnt;
        const float var = s2s[0] * invcnt - mm * mm;
        m[s]  = mm;
        rs[s] = rsqrtf(var + EPS_);
    }
}

// f_e[b,s,c] = (sumN/49 - m[s]) * rs[s]     (mean_N of BN(edge), done analytically)
__global__ void k_fe(const float* __restrict__ sumN, const float* __restrict__ m,
                     const float* __restrict__ rs, float* __restrict__ fe,
                     bf16* __restrict__ feb)
{
    const size_t n = (size_t)B_ * NC_ * NC_ * C_;
    const size_t idx = (size_t)blockIdx.x * blockDim.x + threadIdx.x;
    if (idx >= n) return;
    const int s = (int)((idx / C_) % (NC_ * NC_));
    const float v = (sumN[idx] * (1.0f / N_) - m[s]) * rs[s];
    fe[idx]  = v;
    feb[idx] = (bf16)v;
}

// msg[b,i,j,c] = msgE + Vjx[b,i,c] + Vix[b,j,c]   (in place over msgE)
__global__ void k_msg(const float* __restrict__ vjx, const float* __restrict__ vix,
                      float* __restrict__ msg)
{
    const size_t n = (size_t)B_ * NC_ * NC_ * C_;
    const size_t idx = (size_t)blockIdx.x * blockDim.x + threadIdx.x;
    if (idx >= n) return;
    const int c = (int)(idx % C_);
    const int s = (int)((idx / C_) % (NC_ * NC_));
    const int b = (int)(idx / ((size_t)C_ * NC_ * NC_));
    const int i = s / NC_, j = s % NC_;
    msg[idx] += vjx[((size_t)(b * NC_ + i)) * C_ + c] + vix[((size_t)(b * NC_ + j)) * C_ + c];
}

// edge += relu(bn(msg))  (per-slot stats)
__global__ void k_edge_update(float* __restrict__ fe, bf16* __restrict__ feb,
                              const float* __restrict__ msg, const float* __restrict__ m,
                              const float* __restrict__ rs)
{
    const size_t n = (size_t)B_ * NC_ * NC_ * C_;
    const size_t idx = (size_t)blockIdx.x * blockDim.x + threadIdx.x;
    if (idx >= n) return;
    const int s = (int)((idx / C_) % (NC_ * NC_));
    const float v = fe[idx] + fmaxf(0.0f, (msg[idx] - m[s]) * rs[s]);
    fe[idx]  = v;
    feb[idx] = (bf16)v;
}

// agg[b,j,c] = (1/12) * sum_i softmax_j(sigmoid(edge[b,i,:,c]))[j] * Ujx[b,i,c]
__global__ void k_agg(const float* __restrict__ fe, const float* __restrict__ ujx,
                      float* __restrict__ agg)
{
    const int idx = blockIdx.x * blockDim.x + threadIdx.x;   // over B*C
    if (idx >= B_ * C_) return;
    const int b = idx / C_, c = idx % C_;
    float a[NC_];
#pragma unroll
    for (int j = 0; j < NC_; ++j) a[j] = 0.f;
    for (int i = 0; i < NC_; ++i) {
        float e[NC_];
        float sum = 0.f;
#pragma unroll
        for (int j = 0; j < NC_; ++j) {
            const float x = fe[((size_t)(b * NC_ * NC_ + i * NC_ + j)) * C_ + c];
            const float sg = 1.0f / (1.0f + __expf(-x));
            e[j] = __expf(sg);
            sum += e[j];
        }
        const float u   = ujx[((size_t)(b * NC_ + i)) * C_ + c];
        const float inv = u / sum;
#pragma unroll
        for (int j = 0; j < NC_; ++j) a[j] += e[j] * inv;
    }
#pragma unroll
    for (int j = 0; j < NC_; ++j)
        agg[((size_t)(b * NC_ + j)) * C_ + c] = a[j] * (1.0f / NC_);
}

__global__ void k_xn(const float* __restrict__ ux, const float* __restrict__ agg,
                     float* __restrict__ xn)
{
    const size_t n = (size_t)B_ * NC_ * C_;
    const size_t idx = (size_t)blockIdx.x * blockDim.x + threadIdx.x;
    if (idx >= n) return;
    xn[idx] = ux[idx] + agg[idx];
}

// f_v = relu(f_v + bn(xn))  (per-class stats)
__global__ void k_v_update(float* __restrict__ fv, bf16* __restrict__ fvb,
                           const float* __restrict__ xn, const float* __restrict__ m,
                           const float* __restrict__ rs)
{
    const size_t n = (size_t)B_ * NC_ * C_;
    const size_t idx = (size_t)blockIdx.x * blockDim.x + threadIdx.x;
    if (idx >= n) return;
    const int k = (int)((idx / C_) % NC_);
    const float v = fmaxf(0.0f, fv[idx] + (xn[idx] - m[k]) * rs[k]);
    fv[idx]  = v;
    fvb[idx] = (bf16)v;
}

// gather the 66 upper-triangular class pairs into the MLP-head inputs (bf16)
__global__ void k_gather(const float* __restrict__ fe, const float* __restrict__ fv,
                         bf16* __restrict__ edges, bf16* __restrict__ nodes)
{
    const int bp = blockIdx.x;           // b*66 + p
    const int p  = bp % 66;
    const int b  = bp / 66;
    int i = 0, j = 0, pp = p;
    for (int ii = 0; ii < NC_; ++ii) {
        const int cnt = NC_ - 1 - ii;
        if (pp < cnt) { i = ii; j = ii + 1 + pp; break; }
        pp -= cnt;
    }
    const float* e1 = fe + ((size_t)(b * NC_ * NC_ + i * NC_ + j)) * C_;
    const float* e2 = fe + ((size_t)(b * NC_ * NC_ + j * NC_ + i)) * C_;
    const float* n1 = fv + ((size_t)(b * NC_ + i)) * C_;
    const float* n2 = fv + ((size_t)(b * NC_ + j)) * C_;
    bf16* eo = edges + (size_t)bp * (2 * C_);
    bf16* no = nodes + (size_t)bp * (2 * C_);
    for (int c = threadIdx.x; c < C_; c += blockDim.x) {
        eo[c]       = (bf16)e1[c];
        eo[C_ + c]  = (bf16)e2[c];
        no[c]       = (bf16)n1[c];
        no[C_ + c]  = (bf16)n2[c];
    }
}

// mlp stage 3 (256 -> 4, both heads) + final 8 -> 4 linear
__global__ void __launch_bounds__(64)
k_final(const bf16* __restrict__ z2e, const bf16* __restrict__ z2n,
        const float* __restrict__ w3, const float* __restrict__ b3,
        const float* __restrict__ fw, const float* __restrict__ fb,
        float* __restrict__ out)
{
    __shared__ float cat[8];
    const int bp = blockIdx.x;
    const int t  = threadIdx.x;
    if (t < 8) {
        const int grp = t >> 2;     // 0 = edge head, 1 = node head
        const int q   = t & 3;
        const bf16* z  = (grp == 0 ? z2e : z2n) + (size_t)bp * 256;
        const float* w = w3 + (size_t)grp * 4 * 256 + q * 256;
        float s = b3[grp * 4 + q];
        for (int c = 0; c < 256; ++c) s += (float)z[c] * w[c];
        cat[t] = s;
    }
    __syncthreads();
    if (t < 4) {
        float s = fb[t];
#pragma unroll
        for (int u = 0; u < 8; ++u) s += cat[u] * fw[t * 8 + u];
        out[(size_t)bp * 4 + t] = s;
    }
}

// ---------------------------------------------------------------------------
// Launcher
// ---------------------------------------------------------------------------
extern "C" void kernel_launch(void* const* d_in, const int* in_sizes, int n_in,
                              void* d_out, int out_size, void* d_ws, size_t ws_size,
                              hipStream_t stream)
{
    (void)in_sizes; (void)n_in; (void)out_size; (void)ws_size;

    const float* x      = (const float*)d_in[0];
    const float* cls_w  = (const float*)d_in[1];
    const float* cls_b  = (const float*)d_in[2];
    const float* aqw    = (const float*)d_in[3];
    const float* aqb    = (const float*)d_in[4];
    const float* akw    = (const float*)d_in[5];
    const float* akb    = (const float*)d_in[6];
    const float* avw    = (const float*)d_in[7];
    const float* avb    = (const float*)d_in[8];
    const float* ep_w   = (const float*)d_in[9];
    const float* ep_b   = (const float*)d_in[10];
    const float* gnn_w  = (const float*)d_in[11];
    const float* mw1    = (const float*)d_in[12];
    const float* mb1    = (const float*)d_in[13];
    const float* mw2    = (const float*)d_in[14];
    const float* mb2    = (const float*)d_in[15];
    const float* mw3    = (const float*)d_in[16];
    const float* mb3    = (const float*)d_in[17];
    const float* fw     = (const float*)d_in[18];
    const float* fb     = (const float*)d_in[19];
    float* out          = (float*)d_out;

    // ---- workspace carve-up -------------------------------------------------
    char*  wsp = (char*)d_ws;
    size_t off = 0;
    auto alloc = [&](size_t bytes) -> void* {
        void* p = wsp + off;
        off += (bytes + 255) & ~(size_t)255;
        return p;
    };
    const size_t W512 = (size_t)C_ * C_;                 // 262144
    bf16* x_bf    = (bf16*)alloc((size_t)B_ * N_ * C_ * 2);
    bf16* clsw_bf = (bf16*)alloc((size_t)NC_ * W512 * 2);
    bf16* qw_bf   = (bf16*)alloc(2 * (size_t)256 * C_ * 2);
    bf16* kw_bf   = (bf16*)alloc(2 * (size_t)256 * C_ * 2);
    bf16* vw_bf   = (bf16*)alloc(2 * W512 * 2);
    bf16* epw_bf  = (bf16*)alloc(W512 * 2);
    bf16* gnnw_bf = (bf16*)alloc(10 * W512 * 2);
    bf16* w1_bf   = (bf16*)alloc(2 * (size_t)C_ * 1024 * 2);
    bf16* w2_bf   = (bf16*)alloc(2 * (size_t)256 * C_ * 2);

    const size_t nH  = (size_t)B_ * N_ * NC_ * C_;       // 4,816,896
    float* h1    = (float*)alloc(nH * 4);
    float* bn_m  = (float*)alloc((size_t)NC_ * C_ * 4);
    float* bn_r  = (float*)alloc((size_t)NC_ * C_ * 4);
    bf16*  fu_bf = (bf16*)alloc(nH * 2);
    float* fv    = (float*)alloc((size_t)B_ * NC_ * C_ * 4);
    bf16*  fv_bf = (bf16*)alloc((size_t)B_ * NC_ * C_ * 2);
    float* q1    = (float*)alloc((size_t)B_ * N_ * NC_ * 256 * 4);
    float* k1    = (float*)alloc((size_t)B_ * N_ * 256 * 4);
    float* v1    = (float*)alloc((size_t)B_ * N_ * C_ * 4);
    bf16*  feat_bf = (bf16*)alloc(nH * 2);
    float* q2    = (float*)alloc((size_t)B_ * NC_ * N_ * 256 * 4);
    float* k2    = (float*)alloc((size_t)B_ * NC_ * N_ * 256 * 4);
    float* v2    = (float*)alloc(nH * 4);
    const size_t nF2 = (size_t)B_ * NC_ * NC_ * N_ * C_; // 57,802,752
    bf16*  feat2_bf = (bf16*)alloc(nF2 * 2);
    bf16*  h_bf     = (bf16*)alloc(nF2 * 2);
    float* sumN  = (float*)alloc((size_t)B_ * NC_ * NC_ * C_ * 4);
    float* sm    = (float*)alloc(144 * 4);
    float* sr    = (float*)alloc(144 * 4);
    const size_t nE = (size_t)B_ * NC_ * NC_ * C_;       // 1,179,648
    float* fe    = (float*)alloc(nE * 4);
    bf16*  fe_bf = (bf16*)alloc(nE * 2);
    float* vix   = (float*)alloc((size_t)B_ * NC_ * C_ * 4);
    float* vjx   = (float*)alloc((size_t)B_ * NC_ * C_ * 4);
    float* ujx   = (float*)alloc((size_t)B_ * NC_ * C_ * 4);
    float* ux    = (float*)alloc((size_t)B_ * NC_ * C_ * 4);
    float* msg   = (float*)alloc(nE * 4);
    float* gm    = (float*)alloc(144 * 4);
    float* gr    = (float*)alloc(144 * 4);
    float* agg   = (float*)alloc((size_t)B_ * NC_ * C_ * 4);
    float* xn    = (float*)alloc((size_t)B_ * NC_ * C_ * 4);
    float* xm    = (float*)alloc(NC_ * 4);
    float* xr    = (float*)alloc(NC_ * 4);
    bf16* edges_bf = (bf16*)alloc((size_t)B_ * 66 * 1024 * 2);
    bf16* nodes_bf = (bf16*)alloc((size_t)B_ * 66 * 1024 * 2);
    bf16* z1e_bf = (bf16*)alloc((size_t)B_ * 66 * 512 * 2);
    bf16* z1n_bf = (bf16*)alloc((size_t)B_ * 66 * 512 * 2);
    bf16* z2e_bf = (bf16*)alloc((size_t)B_ * 66 * 256 * 2);
    bf16* z2n_bf = (bf16*)alloc((size_t)B_ * 66 * 256 * 2);

    auto cvt = [&](const float* s, bf16* d, size_t n) {
        const int blocks = (int)(((n + 255) / 256) < 4096 ? ((n + 255) / 256) : 4096);
        k_cvt<<<blocks, 256, 0, stream>>>(s, d, n);
    };
    // 32x64 macro-tiles: grid = ceil(M/32) * (N/64)
    auto gemm = [&](const bf16* A, const bf16* W, const float* bias,
                    float* Cf, bf16* Cb, int M, int N, int K, int relu) {
        const int tiles = ((M + 31) / 32) * (N / 64);
        k_gemm_wmma<<<tiles, 32, 0, stream>>>(A, W, bias, Cf, Cb, M, N, K, relu);
    };

    // ---- bf16 conversions ---------------------------------------------------
    cvt(x, x_bf, (size_t)B_ * N_ * C_);
    cvt(cls_w, clsw_bf, (size_t)NC_ * W512);
    cvt(aqw, qw_bf, 2 * (size_t)256 * C_);
    cvt(akw, kw_bf, 2 * (size_t)256 * C_);
    cvt(avw, vw_bf, 2 * W512);
    cvt(ep_w, epw_bf, W512);
    cvt(gnn_w, gnnw_bf, 10 * W512);
    cvt(mw1, w1_bf, 2 * (size_t)C_ * 1024);
    cvt(mw2, w2_bf, 2 * (size_t)256 * C_);

    // ---- stage 1: per-class projection + BN + ReLU + f_v --------------------
    // h1(784 x 6144) = x(784x512) @ cls_w^T(512x6144) + cls_b  (cols = (k,o))
    gemm(x_bf, clsw_bf, cls_b, h1, nullptr, B_ * N_, NC_ * C_, C_, 0);
    k_col_stats<<<(NC_ * C_ + 255) / 256, 256, 0, stream>>>(h1, bn_m, bn_r, B_ * N_, NC_ * C_);
    k_bn_relu_bf16<<<(int)((nH + 255) / 256), 256, 0, stream>>>(h1, bn_m, bn_r, fu_bf, nH, NC_ * C_);
    k_fv<<<B_ * NC_, 256, 0, stream>>>(fu_bf, fv, fv_bf);

    // ---- stage 2: cross-attn 1 (FAM) ---------------------------------------
    gemm(fu_bf, qw_bf,            aqb,       q1, nullptr, B_ * N_ * NC_, 256, C_, 0);
    gemm(x_bf,  kw_bf,            akb,       k1, nullptr, B_ * N_,       256, C_, 0);
    gemm(x_bf,  vw_bf,            avb,       v1, nullptr, B_ * N_,       C_,  C_, 0);
    k_attn1<<<B_ * NC_ * N_, 128, 0, stream>>>(q1, k1, v1, feat_bf);

    // ---- stage 3: cross-attn 2 (ARM) over all 144 class pairs ---------------
    gemm(feat_bf, qw_bf + (size_t)256 * C_, aqb + 256, q2, nullptr, B_ * NC_ * N_, 256, C_, 0);
    gemm(feat_bf, kw_bf + (size_t)256 * C_, akb + 256, k2, nullptr, B_ * NC_ * N_, 256, C_, 0);
    gemm(feat_bf, vw_bf + W512,             avb + 512, v2, nullptr, B_ * NC_ * N_, C_,  C_, 0);
    k_attn2<<<B_ * NC_ * NC_ * N_, 128, 0, stream>>>(q2, k2, v2, feat2_bf);

    // ---- stage 4: edge projection + BN + mean over N ------------------------
    gemm(feat2_bf, epw_bf, ep_b, nullptr, h_bf, B_ * NC_ * NC_ * N_, C_, C_, 0);
    k_sumN<<<B_ * NC_ * NC_, 256, 0, stream>>>(h_bf, sumN);
    k_slot_stats<bf16><<<NC_ * NC_, 256, 0, stream>>>(
        h_bf, sm, sr, B_, (size_t)NC_ * NC_ * N_ * C_, (size_t)N_ * C_, N_ * C_,
        1.0f / ((float)B_ * N_ * C_));
    k_fe<<<(int)((nE + 255) / 256), 256, 0, stream>>>(sumN, sm, sr, fe, fe_bf);

    // ---- stage 5: two GCN message-passing blocks ----------------------------
    for (int blk = 0; blk < 2; ++blk) {
        const bf16* wb = gnnw_bf + (size_t)blk * 5 * W512;  // U,V,A,B,E
        gemm(fv_bf, wb + 0 * W512, nullptr, ux,  nullptr, B_ * NC_, C_, C_, 0);  // x@U^T
        gemm(fv_bf, wb + 1 * W512, nullptr, ujx, nullptr, B_ * NC_, C_, C_, 0);  // x@V^T
        gemm(fv_bf, wb + 2 * W512, nullptr, vix, nullptr, B_ * NC_, C_, C_, 0);  // x@A^T
        gemm(fv_bf, wb + 3 * W512, nullptr, vjx, nullptr, B_ * NC_, C_, C_, 0);  // x@B^T
        gemm(fe_bf, wb + 4 * W512, nullptr, msg, nullptr, B_ * NC_ * NC_, C_, C_, 0); // edge@E^T
        k_msg<<<(int)((nE + 255) / 256), 256, 0, stream>>>(vjx, vix, msg);
        k_slot_stats<float><<<NC_ * NC_, 256, 0, stream>>>(
            msg, gm, gr, B_, (size_t)NC_ * NC_ * C_, (size_t)C_, C_,
            1.0f / ((float)B_ * C_));
        k_edge_update<<<(int)((nE + 255) / 256), 256, 0, stream>>>(fe, fe_bf, msg, gm, gr);
        k_agg<<<(B_ * C_ + 255) / 256, 256, 0, stream>>>(fe, ujx, agg);
        k_xn<<<(B_ * NC_ * C_ + 255) / 256, 256, 0, stream>>>(ux, agg, xn);
        k_slot_stats<float><<<NC_, 256, 0, stream>>>(
            xn, xm, xr, B_, (size_t)NC_ * C_, (size_t)C_, C_,
            1.0f / ((float)B_ * C_));
        k_v_update<<<(B_ * NC_ * C_ + 255) / 256, 256, 0, stream>>>(fv, fv_bf, xn, xm, xr);
    }

    // ---- stage 6: pairwise class-link head ----------------------------------
    k_gather<<<B_ * 66, 128, 0, stream>>>(fe, fv, edges_bf, nodes_bf);
    gemm(edges_bf, w1_bf,                     mb1,       nullptr, z1e_bf, B_ * 66, C_,  1024, 1);
    gemm(nodes_bf, w1_bf + (size_t)C_ * 1024, mb1 + C_,  nullptr, z1n_bf, B_ * 66, C_,  1024, 1);
    gemm(z1e_bf,   w2_bf,                     mb2,       nullptr, z2e_bf, B_ * 66, 256, C_,   1);
    gemm(z1n_bf,   w2_bf + (size_t)256 * C_,  mb2 + 256, nullptr, z2n_bf, B_ * 66, 256, C_,   1);
    k_final<<<B_ * 66, 64, 0, stream>>>(z2e_bf, z2n_bf, mw3, mb3, fw, fb, out);
}
